// Attention_78151224918608
// MI455X (gfx1250) — matrix operand order repaired and checked by reading, not compiled, not run on hardware
//
#include <hip/hip_runtime.h>
#include <hip/hip_bf16.h>

typedef __attribute__((ext_vector_type(16))) __bf16 v16bf;
typedef __attribute__((ext_vector_type(8)))  __bf16 bf16x8;
typedef __attribute__((ext_vector_type(8)))  float  v8f;

#define DIMM 2048
#define NH   32
#define NKV  8
#define HD   64
#define BB   2
#define SS   2048

union AFrag { v16bf v; bf16x8 h[2]; };

// ---------------- elementwise conversion / layout kernels ----------------

__global__ void cvt_f32_bf16(const float* __restrict__ src, __bf16* __restrict__ dst, int n) {
  int i = blockIdx.x * blockDim.x + threadIdx.x;
  if (i < n) dst[i] = (__bf16)src[i];
}

// W is K x N (row major, f32) -> Wt is N x K (row major, bf16)
__global__ void transpose_cvt(const float* __restrict__ W, __bf16* __restrict__ Wt, int K, int N) {
  int i = blockIdx.x * blockDim.x + threadIdx.x;
  if (i >= K * N) return;
  int k = i / N, n = i - k * N;
  Wt[(size_t)n * K + k] = (__bf16)W[i];
}

// Qtmp [B,S,NH*HD] bf16 -> RoPE -> Q [B,NH,S,HD]
__global__ void rope_q_kernel(const __bf16* __restrict__ T, const float* __restrict__ fc,
                              const float* __restrict__ fs, __bf16* __restrict__ Out) {
  int i = blockIdx.x * blockDim.x + threadIdx.x;
  int d2 = i & 31; int t = i >> 5;
  int h  = t & (NH - 1); t >>= 5;
  int s  = t & (SS - 1); int b = t >> 11;
  size_t src = ((size_t)b * SS + s) * DIMM + h * HD + 2 * d2;
  float tr = (float)T[src], ti = (float)T[src + 1];
  float c = fc[s * 32 + d2], sn = fs[s * 32 + d2];
  size_t dst = (((size_t)b * NH + h) * SS + s) * HD + 2 * d2;
  Out[dst]     = (__bf16)(tr * c - ti * sn);
  Out[dst + 1] = (__bf16)(tr * sn + ti * c);
}

// Ktmp [B,S,NKV*HD] bf16 -> RoPE -> K [B,NKV,S,HD]
__global__ void rope_k_kernel(const __bf16* __restrict__ T, const float* __restrict__ fc,
                              const float* __restrict__ fs, __bf16* __restrict__ Out) {
  int i = blockIdx.x * blockDim.x + threadIdx.x;
  int d2 = i & 31; int t = i >> 5;
  int h  = t & (NKV - 1); t >>= 3;
  int s  = t & (SS - 1); int b = t >> 11;
  size_t src = ((size_t)b * SS + s) * (NKV * HD) + h * HD + 2 * d2;
  float tr = (float)T[src], ti = (float)T[src + 1];
  float c = fc[s * 32 + d2], sn = fs[s * 32 + d2];
  size_t dst = (((size_t)b * NKV + h) * SS + s) * HD + 2 * d2;
  Out[dst]     = (__bf16)(tr * c - ti * sn);
  Out[dst + 1] = (__bf16)(tr * sn + ti * c);
}

// Vtmp [B,S,NKV*HD] -> Vt [B,NKV,HD,S]  (transposed so P@V B-frags are contiguous)
__global__ void transpose_v_kernel(const __bf16* __restrict__ T, __bf16* __restrict__ Out) {
  int i = blockIdx.x * blockDim.x + threadIdx.x;
  int d  = i & 63; int t = i >> 6;
  int kv = t & (NKV - 1); t >>= 3;
  int s  = t & (SS - 1); int b = t >> 11;
  size_t src = ((size_t)b * SS + s) * (NKV * HD) + kv * HD + d;
  size_t dst = (((size_t)b * NKV + kv) * HD + d) * SS + s;
  Out[dst] = T[src];
}

// ---------------- WMMA bf16 GEMM:  C[M,N] = A[M,K] * Bt[N,K]^T ----------------
// 256 threads = 8 waves (4 in M x 2 in N). Wave tile = 16(M) x 64(N): one A
// fragment is reused across 4 B fragments / 4 WMMAs per k-step (2.5 b128 loads
// per WMMA instead of 4). Block tile = 64 x 128.
__global__ void gemm_bf16_tn(const __bf16* __restrict__ A, const __bf16* __restrict__ Bt,
                             __bf16* __restrict__ C, int M, int N, int K) {
  int lane = threadIdx.x & 31, wave = threadIdx.x >> 5;
  int wm = wave & 3, wn = wave >> 2;
  int mbase = blockIdx.x * 64 + wm * 16;
  int nbase = blockIdx.y * 128 + wn * 64;
  int l16 = lane & 15, half = lane >> 4;
  const __bf16* arow = A + (size_t)(mbase + l16) * K;
  const __bf16* brow0 = Bt + (size_t)(nbase + l16) * K;        // n-subtile 0
  const __bf16* brow1 = brow0 + (size_t)16 * K;                // n-subtile 1
  const __bf16* brow2 = brow0 + (size_t)32 * K;                // n-subtile 2
  const __bf16* brow3 = brow0 + (size_t)48 * K;                // n-subtile 3
  v8f acc0 = {}, acc1 = {}, acc2 = {}, acc3 = {};
  for (int kt = 0; kt < K; kt += 32) {
    AFrag a;
    a.h[0] = *(const bf16x8*)(arow + kt + 16 * half);          // k = 16*half + 0..7
    a.h[1] = *(const bf16x8*)(arow + kt + 16 * half + 8);      // k = 16*half + 8..15
    v16bf b0 = *(const v16bf*)(brow0 + kt + 16 * half);        // k = 16*half + 0..15
    v16bf b1 = *(const v16bf*)(brow1 + kt + 16 * half);
    v16bf b2 = *(const v16bf*)(brow2 + kt + 16 * half);
    v16bf b3 = *(const v16bf*)(brow3 + kt + 16 * half);
    acc0 = __builtin_amdgcn_wmma_f32_16x16x32_bf16(false, a.v, false, b0, (short)0, acc0, false, false);
    acc1 = __builtin_amdgcn_wmma_f32_16x16x32_bf16(false, a.v, false, b1, (short)0, acc1, false, false);
    acc2 = __builtin_amdgcn_wmma_f32_16x16x32_bf16(false, a.v, false, b2, (short)0, acc2, false, false);
    acc3 = __builtin_amdgcn_wmma_f32_16x16x32_bf16(false, a.v, false, b3, (short)0, acc3, false, false);
  }
#pragma unroll
  for (int v = 0; v < 8; ++v) {
    size_t row = (size_t)(mbase + v + 8 * half) * N + nbase + l16;
    C[row]      = (__bf16)acc0[v];
    C[row + 16] = (__bf16)acc1[v];
    C[row + 32] = (__bf16)acc2[v];
    C[row + 48] = (__bf16)acc3[v];
  }
}

// ---------------- Flash attention, one 16-row q-tile per wave ----------------
__global__ void attn_fmha(const __bf16* __restrict__ Q, const __bf16* __restrict__ Kk,
                          const __bf16* __restrict__ Vt, __bf16* __restrict__ O) {
  __shared__ __bf16 plds[8][16][32];   // per-wave private P tile (no cross-wave sync needed)
  int lane = threadIdx.x & 31, wave = threadIdx.x >> 5;
  int id = blockIdx.x * 8 + wave;      // over B * NH * (S/16)
  int qtile = id & 127;
  int hh = (id >> 7) & 31;
  int b  = id >> 12;
  int kvh = hh >> 2;                   // N_REP = 4
  int qbase = qtile << 4;
  int l16 = lane & 15, half = lane >> 4;

  // Q A-fragments for d = 0..31 and 32..63
  const __bf16* qrow = Q + (((size_t)b * NH + hh) * SS + qbase + l16) * HD;
  AFrag aq0, aq1;
  aq0.h[0] = *(const bf16x8*)(qrow + 16 * half);
  aq0.h[1] = *(const bf16x8*)(qrow + 8 + 16 * half);
  aq1.h[0] = *(const bf16x8*)(qrow + 32 + 16 * half);
  aq1.h[1] = *(const bf16x8*)(qrow + 40 + 16 * half);

  const __bf16* kbase = Kk + ((size_t)b * NKV + kvh) * SS * HD;
  const __bf16* vbase = Vt + ((size_t)b * NKV + kvh) * HD * SS;

  v8f oacc[4] = {};
  float rmax[8], rsum[8];
#pragma unroll
  for (int v = 0; v < 8; ++v) { rmax[v] = -3.0e38f; rsum[v] = 0.f; }

  const float scale = 0.125f;                 // 1/sqrt(64)
  int nkt = (qbase + 47) >> 5;                // 32-key tiles covering keys <= qbase+15
  for (int kt = 0; kt < nkt; ++kt) {
    int kb = kt << 5;
    // K^T B-fragments: lane column = key (kb + 16t + l16), rows = d (contiguous)
    const __bf16* krow0 = kbase + (size_t)(kb + l16) * HD;
    const __bf16* krow1 = krow0 + 16 * HD;
    v16bf bk00 = *(const v16bf*)(krow0 + 16 * half);
    v16bf bk01 = *(const v16bf*)(krow0 + 32 + 16 * half);
    v16bf bk10 = *(const v16bf*)(krow1 + 16 * half);
    v16bf bk11 = *(const v16bf*)(krow1 + 32 + 16 * half);
    v8f c0 = {}, c1 = {};
    c0 = __builtin_amdgcn_wmma_f32_16x16x32_bf16(false, aq0.v, false, bk00, (short)0, c0, false, false);
    c0 = __builtin_amdgcn_wmma_f32_16x16x32_bf16(false, aq1.v, false, bk01, (short)0, c0, false, false);
    c1 = __builtin_amdgcn_wmma_f32_16x16x32_bf16(false, aq0.v, false, bk10, (short)0, c1, false, false);
    c1 = __builtin_amdgcn_wmma_f32_16x16x32_bf16(false, aq1.v, false, bk11, (short)0, c1, false, false);

    // online softmax per row (row m = v + 8*half lives across 16 lanes of one half-wave)
#pragma unroll
    for (int v = 0; v < 8; ++v) {
      int qidx = qbase + v + 8 * half;
      float s0 = c0[v] * scale;
      float s1 = c1[v] * scale;
      bool m0 = (kb + l16) > qidx;
      bool m1 = (kb + 16 + l16) > qidx;
      float mx = fmaxf(m0 ? -3.0e38f : s0, m1 ? -3.0e38f : s1);
      for (int off = 1; off < 16; off <<= 1)
        mx = fmaxf(mx, __shfl_xor(mx, off, 32));
      float nm = fmaxf(rmax[v], mx);
      float alpha = __expf(rmax[v] - nm);
      float p0 = m0 ? 0.f : __expf(s0 - nm);
      float p1 = m1 ? 0.f : __expf(s1 - nm);
      float ts = p0 + p1;
      for (int off = 1; off < 16; off <<= 1)
        ts += __shfl_xor(ts, off, 32);
      rsum[v] = rsum[v] * alpha + ts;
      rmax[v] = nm;
#pragma unroll
      for (int j = 0; j < 4; ++j) oacc[j][v] *= alpha;
      int mrow = v + 8 * half;
      plds[wave][mrow][l16]      = (__bf16)p0;
      plds[wave][mrow][16 + l16] = (__bf16)p1;
    }

    // Reload P as A-fragment (C-frag -> A-frag transpose via LDS; same-wave DS ops are in-order)
    AFrag ap;
    ap.h[0] = *(const bf16x8*)&plds[wave][l16][16 * half];
    ap.h[1] = *(const bf16x8*)&plds[wave][l16][8 + 16 * half];
    // V^T B-fragments: lane column = d (16j + l16), rows = keys (contiguous in Vt)
#pragma unroll
    for (int j = 0; j < 4; ++j) {
      v16bf bv = *(const v16bf*)(vbase + (size_t)(16 * j + l16) * SS + kb + 16 * half);
      oacc[j] = __builtin_amdgcn_wmma_f32_16x16x32_bf16(false, ap.v, false, bv,
                                                        (short)0, oacc[j], false, false);
    }
  }

  float inv[8];
#pragma unroll
  for (int v = 0; v < 8; ++v) inv[v] = 1.0f / rsum[v];
#pragma unroll
  for (int j = 0; j < 4; ++j)
#pragma unroll
    for (int v = 0; v < 8; ++v) {
      size_t off = ((size_t)b * SS + qbase + v + 8 * half) * DIMM + hh * HD + 16 * j + l16;
      O[off] = (__bf16)(oacc[j][v] * inv[v]);
    }
}

// ---------------- launcher ----------------

extern "C" void kernel_launch(void* const* d_in, const int* in_sizes, int n_in,
                              void* d_out, int out_size, void* d_ws, size_t ws_size,
                              hipStream_t stream) {
  (void)in_sizes; (void)n_in; (void)out_size; (void)ws_size;
  const float* x    = (const float*)d_in[0];
  const float* fcos = (const float*)d_in[1];
  const float* fsin = (const float*)d_in[2];
  const float* wq   = (const float*)d_in[3];
  const float* wk   = (const float*)d_in[4];
  const float* wv   = (const float*)d_in[5];
  const float* wo   = (const float*)d_in[6];

  char* ws = (char*)d_ws;
  const size_t MB = 1ull << 20;
  __bf16* Xbf  = (__bf16*)(ws + 0);         // [4096,2048]      16MB
  __bf16* WqT  = (__bf16*)(ws + 16 * MB);   // [2048,2048]       8MB
  __bf16* WkT  = (__bf16*)(ws + 24 * MB);   // [512,2048]        2MB
  __bf16* WvT  = (__bf16*)(ws + 26 * MB);   // [512,2048]        2MB
  __bf16* WoT  = (__bf16*)(ws + 28 * MB);   // [2048,2048]       8MB
  __bf16* Qt   = (__bf16*)(ws + 36 * MB);   // [B,S,2048]       16MB
  __bf16* Kt   = (__bf16*)(ws + 52 * MB);   // [B,S,512]         4MB
  __bf16* Vtm  = (__bf16*)(ws + 56 * MB);   // [B,S,512]         4MB
  __bf16* Qr   = (__bf16*)(ws + 60 * MB);   // [B,NH,S,64]      16MB
  __bf16* Kr   = (__bf16*)(ws + 76 * MB);   // [B,NKV,S,64]      4MB
  __bf16* Vt   = (__bf16*)(ws + 80 * MB);   // [B,NKV,64,S]      4MB
  __bf16* Attn = (__bf16*)(ws + 84 * MB);   // [B,S,2048]       16MB

  cvt_f32_bf16<<<(BB * SS * DIMM) / 256, 256, 0, stream>>>(x, Xbf, BB * SS * DIMM);
  transpose_cvt<<<(DIMM * DIMM) / 256, 256, 0, stream>>>(wq, WqT, DIMM, DIMM);
  transpose_cvt<<<(DIMM * 512) / 256, 256, 0, stream>>>(wk, WkT, DIMM, 512);
  transpose_cvt<<<(DIMM * 512) / 256, 256, 0, stream>>>(wv, WvT, DIMM, 512);
  transpose_cvt<<<(DIMM * DIMM) / 256, 256, 0, stream>>>(wo, WoT, DIMM, DIMM);

  gemm_bf16_tn<<<dim3(64, 16), 256, 0, stream>>>(Xbf, WqT, Qt, BB * SS, DIMM, DIMM);
  gemm_bf16_tn<<<dim3(64, 4), 256, 0, stream>>>(Xbf, WkT, Kt, BB * SS, 512, DIMM);
  gemm_bf16_tn<<<dim3(64, 4), 256, 0, stream>>>(Xbf, WvT, Vtm, BB * SS, 512, DIMM);

  rope_q_kernel<<<(BB * SS * NH * 32) / 256, 256, 0, stream>>>(Qt, fcos, fsin, Qr);
  rope_k_kernel<<<(BB * SS * NKV * 32) / 256, 256, 0, stream>>>(Kt, fcos, fsin, Kr);
  transpose_v_kernel<<<(BB * SS * NKV * HD) / 256, 256, 0, stream>>>(Vtm, Vt);

  attn_fmha<<<(BB * NH * (SS / 16)) / 8, 256, 0, stream>>>(Qr, Kr, Vt, Attn);

  gemm_bf16_tn<<<dim3(64, 16), 256, 0, stream>>>(Attn, WoT, (__bf16*)d_out, BB * SS, DIMM, DIMM);
}
